// ModulatedFourierFeatures_30751965839412
// MI455X (gfx1250) — matrix operand order repaired and checked by reading, not compiled, not run on hardware
//
#include <hip/hip_runtime.h>
#include <hip/hip_bf16.h>
#include <math.h>
#include <stdint.h>

#define B_   128
#define T_   1024
#define TM   128          // tokens per block
#define SA   456          // ldsA row stride (elements, 912B: 16B-aligned, bank-spread)
#define SW1  200          // ldsW1 column stride (400B, 8B-aligned)
#define SW2  420          // ldsW2 column stride (840B, 8B-aligned)
#define SX1  264          // x1 tile row stride (528B, 16B-aligned)

typedef __attribute__((ext_vector_type(16))) __bf16 v16bf;
typedef __attribute__((ext_vector_type(8)))  float  v8f;

struct __align__(16) U4 { unsigned int x, y, z, w; };
struct __align__(8)  U2 { unsigned int x, y; };

union Frag {
  v16bf v;
  U4 q[2];
  U2 d[4];
};

// software RNE (host-side pack kernels, not perf critical)
__device__ __forceinline__ unsigned short f2bf(float f) {
  unsigned int x = __float_as_uint(f);
  x += 0x7FFFu + ((x >> 16) & 1u);
  return (unsigned short)(x >> 16);
}
// native convert (main kernel hot paths)
__device__ __forceinline__ unsigned short f2bfn(float f) {
  return __builtin_bit_cast(unsigned short, (__bf16)f);
}
__device__ __forceinline__ float bf2f(unsigned short u) {
  return __uint_as_float(((unsigned int)u) << 16);
}
__device__ __forceinline__ unsigned ldsOff(const void* p) {
  return (unsigned)(uintptr_t)p;   // low 32 bits of flat addr == LDS byte offset
}
// async copy 16B global -> LDS (ASYNCcnt-tracked, saddr + 32-bit voffset form)
__device__ __forceinline__ void asyncCopy16(unsigned ldsDst, unsigned gOff,
                                            const void* sBase) {
  asm volatile("global_load_async_to_lds_b128 %0, %1, %2"
               :: "v"(ldsDst), "v"(gOff), "s"(sBase) : "memory");
}
__device__ __forceinline__ void asyncWait0() {
  asm volatile("s_wait_asynccnt 0x0" ::: "memory");
}

// ---------------------------------------------------------------------------
// Pack fused weight panels (bf16, N-major: row = output column, K contiguous)
//   W1p (256 x 192): [ ltm_w[128:256,:256] ; ltm_w[256:288,:256] ; l0_w ; 0pad16 ]
//   W2p (256 x 416): [ ltm_w[128:256,256:] ; ltm_w[256:288,256:] ; l1_w ]
// ---------------------------------------------------------------------------
__global__ void pack_weights(const float* __restrict__ ltm_w,
                             const float* __restrict__ l0_w,
                             const float* __restrict__ l1_w,
                             unsigned short* __restrict__ wsW1,
                             unsigned short* __restrict__ wsW2) {
  int idx = blockIdx.x * 256 + threadIdx.x;
  if (idx < 256 * 192) {
    int n = idx / 192, k = idx % 192;
    float v;
    if (k < 128)      v = ltm_w[(128 + k) * 512 + n];
    else if (k < 160) v = ltm_w[(256 + (k - 128)) * 512 + n];
    else if (k < 176) v = l0_w[(k - 160) * 256 + n];
    else              v = 0.0f;
    wsW1[n * 192 + k] = f2bf(v);
  } else {
    int j = idx - 256 * 192;
    if (j < 256 * 416) {
      int n = j / 416, k = j % 416;
      float v;
      if (k < 128)      v = ltm_w[(128 + k) * 512 + 256 + n];
      else if (k < 160) v = ltm_w[(256 + (k - 128)) * 512 + 256 + n];
      else              v = l1_w[(k - 160) * 256 + n];
      wsW2[n * 416 + k] = f2bf(v);
    }
  }
}

// ---------------------------------------------------------------------------
// Per-batch bias:  bias0/1[b][n] = ltm_b + layer bias + code@Wc + dir@Wd
// ---------------------------------------------------------------------------
__global__ void bias_kernel(const float* __restrict__ code,
                            const int*   __restrict__ dir_idx,
                            const float* __restrict__ dir_emb,
                            const float* __restrict__ ltm_w,
                            const float* __restrict__ ltm_b,
                            const float* __restrict__ l0_b,
                            const float* __restrict__ l1_b,
                            float* __restrict__ bias0,
                            float* __restrict__ bias1) {
  int b = blockIdx.x, n = threadIdx.x;
  float s0 = ltm_b[n] + l0_b[n];
  float s1 = ltm_b[256 + n] + l1_b[n];
  for (int k = 0; k < 128; ++k) {
    float c = code[b * 128 + k];
    s0 += c * ltm_w[k * 512 + n];
    s1 += c * ltm_w[k * 512 + 256 + n];
  }
  int d = dir_idx[b];
  for (int j = 0; j < 8; ++j) {
    float dv = dir_emb[d * 8 + j];
    s0 += dv * ltm_w[(288 + j) * 512 + n];
    s1 += dv * ltm_w[(288 + j) * 512 + 256 + n];
  }
  bias0[b * 256 + n] = s0;
  bias1[b * 256 + n] = s1;
}

// ---------------------------------------------------------------------------
// Main fused kernel: one block = 128 tokens of one batch.
// A-panel K layout: [hs 0:128 | spatial 128:160 | pos 160:192 | x0 192:448]
// ---------------------------------------------------------------------------
__global__ void __launch_bounds__(256)
mff_main(const float* __restrict__ coords,
         const float* __restrict__ hs,
         const float* __restrict__ x_statics,
         const float* __restrict__ se_w1, const float* __restrict__ se_b1,
         const float* __restrict__ se_w2, const float* __restrict__ se_b2,
         const unsigned short* __restrict__ wsW1,
         const unsigned short* __restrict__ wsW2,
         const float* __restrict__ bias0g, const float* __restrict__ bias1g,
         const float* __restrict__ l2_w, const float* __restrict__ l2_b,
         float* __restrict__ out) {
  __shared__ __align__(16) unsigned short ldsA[TM * SA];     // 116736 B
  __shared__ __align__(16) unsigned short ldsW1[256 * SW1];  // 102400 B (reused: x1|l2w|bias1)
  __shared__ __align__(16) unsigned short ldsW2[128 * SW2];  // 107520 B (head reused: bias0)

  const int tid      = threadIdx.x;
  const int b        = blockIdx.x >> 3;
  const int t0       = (blockIdx.x & 7) * TM;
  const int lane     = tid & 31;
  const int wave     = tid >> 5;
  const int l16      = lane & 15;
  const int laneHalf = lane >> 4;

  // ---------------- Phase 1: stage W1 (async) + A inputs + bias0 ----------
  {
    // W1 panel -> LDS, async so it overlaps the VALU-heavy A-panel build
    for (int idx = tid; idx < 6144; idx += 256) {
      int n = idx / 24, k16 = idx % 24;
      asyncCopy16(ldsOff(ldsW1 + n * SW1 + k16 * 8), idx * 16, wsW1);
    }
    // bias0 -> head of (currently idle) ldsW2
    ((float*)ldsW2)[tid] = bias0g[b * 256 + tid];

    // hs -> ldsA[:, 0:128]
    const float* hsB = hs + ((size_t)b * T_ + t0) * 128;
    for (int idx = tid; idx < TM * 32; idx += 256) {
      int row = idx >> 5, k4 = idx & 31;
      float4 v = ((const float4*)(hsB + row * 128))[k4];
      U2 u;
      u.x = (unsigned int)f2bfn(v.x) | ((unsigned int)f2bfn(v.y) << 16);
      u.y = (unsigned int)f2bfn(v.z) | ((unsigned int)f2bfn(v.w) << 16);
      *(U2*)(ldsA + row * SA + k4 * 4) = u;
    }
    // spatial encoder + positional encoding: 2 threads per token
    {
      int i = tid >> 1, h = tid & 1;
      float4 xs = ((const float4*)x_statics)[(size_t)b * T_ + t0 + i];
      float g[32];
      #pragma unroll
      for (int j = 0; j < 32; ++j) {
        float hj = se_b1[j] + xs.x * se_w1[j] + xs.y * se_w1[32 + j]
                 + xs.z * se_w1[64 + j] + xs.w * se_w1[96 + j];
        g[j] = 0.5f * hj * (1.0f + erff(hj * 0.70710678118654752f));
      }
      for (int j2 = 0; j2 < 16; ++j2) {
        int jo = h * 16 + j2;
        float s = se_b2[jo];
        #pragma unroll
        for (int jj = 0; jj < 32; ++jj) s += g[jj] * se_w2[jj * 32 + jo];
        ldsA[i * SA + 128 + jo] = f2bfn(s);
      }
      float cv = coords[(size_t)b * T_ + t0 + i];
      #pragma unroll
      for (int f = 0; f < 8; ++f) {
        float wv = cv * (1.25f * (float)f);
        float p  = h ? cosf(wv) : sinf(wv);
        ldsA[i * SA + 160 + h * 8 + f] = f2bfn(p);
        ldsA[i * SA + 176 + h * 8 + f] = 0;   // zero pad K 176..191
      }
    }
    asyncWait0();
  }
  __syncthreads();

  // ---------------- Phase 2: GEMM1 -> x0 (written back into A-panel) ------
  {
    const float* bias0Lds = (const float*)ldsW2;
    Frag a1[6];
    #pragma unroll
    for (int c = 0; c < 6; ++c) {
      const U4* p = (const U4*)(ldsA + (wave * 16 + l16) * SA + c * 32 + laneHalf * 8);
      a1[c].q[0] = p[0];   // K k0..k0+7
      a1[c].q[1] = p[2];   // K k0+16..k0+23
    }
    #pragma unroll 4
    for (int nt = 0; nt < 16; ++nt) {
      float bv = bias0Lds[nt * 16 + l16];
      v8f acc = {bv, bv, bv, bv, bv, bv, bv, bv};
      #pragma unroll
      for (int c = 0; c < 6; ++c) {
        Frag bf;
        const U2* pb = (const U2*)(ldsW1 + (nt * 16 + l16) * SW1 + c * 32 + laneHalf * 8);
        bf.d[0] = pb[0]; bf.d[1] = pb[1]; bf.d[2] = pb[4]; bf.d[3] = pb[5];
        acc = __builtin_amdgcn_wmma_f32_16x16x32_bf16(false, a1[c].v, false, bf.v,
                                                      (short)0, acc, false, false);
      }
      #pragma unroll
      for (int r = 0; r < 8; ++r) {
        float v = fmaxf(acc[r], 0.0f);
        int row = wave * 16 + r + laneHalf * 8;
        ldsA[row * SA + 192 + nt * 16 + l16] = f2bfn(v);
      }
    }
  }
  __syncthreads();

  // ---------------- Phase 3/4: GEMM2 over two N-halves ----------------
  Frag a2[13];   // K chunks: 0..4 = hs|spatial, skip pos chunk, 6..13 = x0
  #pragma unroll
  for (int kk = 0; kk < 13; ++kk) {
    int ac = (kk < 5) ? kk : kk + 1;
    const U4* p = (const U4*)(ldsA + (wave * 16 + l16) * SA + ac * 32 + laneHalf * 8);
    a2[kk].q[0] = p[0];
    a2[kk].q[1] = p[2];
  }
  float* l2lds    = (float*)(ldsW1 + TM * SX1);        // 256 floats past x1 tile
  float* bias1Lds = (float*)(ldsW1 + TM * SX1 + 512);  // 256 floats after l2w

  for (int hh = 0; hh < 2; ++hh) {
    // stage W2 half hh (128 columns x 416 elems) asynchronously
    for (int idx = tid; idx < 128 * 52; idx += 256) {
      int n = idx / 52, k16 = idx % 52;
      unsigned gOff = (unsigned)(hh * 128 + n) * 832u + (unsigned)k16 * 16u;
      asyncCopy16(ldsOff(ldsW2 + n * SW2 + k16 * 8), gOff, wsW2);
    }
    if (hh == 0) {   // W1 region dead now: stash l2_w + bias1 in its tail
      l2lds[tid]    = l2_w[tid];
      bias1Lds[tid] = bias1g[b * 256 + tid];
    }
    asyncWait0();
    __syncthreads();

    #pragma unroll 2
    for (int ntl = 0; ntl < 8; ++ntl) {
      int nt = hh * 8 + ntl;
      float bv = bias1Lds[nt * 16 + l16];
      v8f acc = {bv, bv, bv, bv, bv, bv, bv, bv};
      #pragma unroll
      for (int kk = 0; kk < 13; ++kk) {
        Frag bf;
        const U2* pb = (const U2*)(ldsW2 + (ntl * 16 + l16) * SW2 + kk * 32 + laneHalf * 8);
        bf.d[0] = pb[0]; bf.d[1] = pb[1]; bf.d[2] = pb[4]; bf.d[3] = pb[5];
        acc = __builtin_amdgcn_wmma_f32_16x16x32_bf16(false, a2[kk].v, false, bf.v,
                                                      (short)0, acc, false, false);
      }
      #pragma unroll
      for (int r = 0; r < 8; ++r) {
        float v = fmaxf(acc[r], 0.0f);
        int row = wave * 16 + r + laneHalf * 8;
        ldsW1[row * SX1 + nt * 16 + l16] = f2bfn(v);   // x1 tile (bf16)
      }
    }
    __syncthreads();
  }

  // ---------------- Phase 5: out = x1 @ l2_w + l2_b ----------------
  {
    int i = tid >> 1, h = tid & 1;
    float sum = 0.0f;
    #pragma unroll 4
    for (int j = 0; j < 16; ++j) {
      U4 q = *(const U4*)(ldsW1 + i * SX1 + h * 128 + j * 8);
      const float* wv = l2lds + h * 128 + j * 8;
      unsigned int ua[4] = {q.x, q.y, q.z, q.w};
      #pragma unroll
      for (int e = 0; e < 4; ++e) {
        sum += bf2f((unsigned short)(ua[e] & 0xFFFFu)) * wv[2 * e];
        sum += bf2f((unsigned short)(ua[e] >> 16))     * wv[2 * e + 1];
      }
    }
    sum += __shfl_xor(sum, 1, 32);
    if (h == 0) out[(size_t)b * T_ + t0 + i] = sum + l2_b[0];
  }
}

// ---------------------------------------------------------------------------
extern "C" void kernel_launch(void* const* d_in, const int* in_sizes, int n_in,
                              void* d_out, int out_size, void* d_ws, size_t ws_size,
                              hipStream_t stream) {
  (void)in_sizes; (void)n_in; (void)out_size; (void)ws_size;
  const float* coords    = (const float*)d_in[0];
  const float* code      = (const float*)d_in[1];
  const float* hs        = (const float*)d_in[2];
  const float* x_statics = (const float*)d_in[3];
  const int*   dir_idx   = (const int*)d_in[4];
  const float* se_w1   = (const float*)d_in[5];
  const float* se_b1   = (const float*)d_in[6];
  const float* se_w2   = (const float*)d_in[7];
  const float* se_b2   = (const float*)d_in[8];
  const float* dir_emb = (const float*)d_in[9];
  const float* ltm_w   = (const float*)d_in[10];
  const float* ltm_b   = (const float*)d_in[11];
  const float* l0_w    = (const float*)d_in[12];
  const float* l0_b    = (const float*)d_in[13];
  const float* l1_w    = (const float*)d_in[14];
  const float* l1_b    = (const float*)d_in[15];
  const float* l2_w    = (const float*)d_in[16];
  const float* l2_b    = (const float*)d_in[17];
  float* out = (float*)d_out;

  char* ws = (char*)d_ws;
  unsigned short* wsW1 = (unsigned short*)(ws);                       //  98304 B
  unsigned short* wsW2 = (unsigned short*)(ws + 98304);               // 212992 B
  float* bias0 = (float*)(ws + 98304 + 212992);                       // 131072 B
  float* bias1 = (float*)(ws + 98304 + 212992 + 131072);              // 131072 B

  pack_weights<<<608, 256, 0, stream>>>(ltm_w, l0_w, l1_w, wsW1, wsW2);
  bias_kernel<<<128, 256, 0, stream>>>(code, dir_idx, dir_emb, ltm_w, ltm_b,
                                       l0_b, l1_b, bias0, bias1);
  mff_main<<<1024, 256, 0, stream>>>(coords, hs, x_statics,
                                     se_w1, se_b1, se_w2, se_b2,
                                     wsW1, wsW2, bias0, bias1,
                                     l2_w, l2_b, out);
}
